// Actor_28183575396972
// MI455X (gfx1250) — matrix operand order repaired and checked by reading, not compiled, not run on hardware
//
#include <hip/hip_runtime.h>
#include <hip/hip_bf16.h>

#define NNODES 64000
#define NBATCH 128
#define NPG    500
#define INDIM  8
#define DH     128
#define NEDGE  512000
#define BN_EPS 1e-5f

typedef _Float16 f16;
typedef _Float16 v16h __attribute__((ext_vector_type(16)));
typedef _Float16 v8h  __attribute__((ext_vector_type(8)));
typedef float    v8f  __attribute__((ext_vector_type(8)));

static __device__ __forceinline__ v16h cat8(v8h lo, v8h hi) {
  return __builtin_shufflevector(lo, hi, 0,1,2,3,4,5,6,7,8,9,10,11,12,13,14,15);
}

// ---- async global->LDS copy (gfx1250 ISA path via inline asm) ----
// global_load_async_to_lds_b128 vdst(LDS byte addr), vaddr(64b global), off
#define HAVE_ASYNC 1
static __device__ __forceinline__ void async_cp16(const f16* g, f16* l) {
  unsigned lds = (unsigned)(unsigned long long)l;  // low 32 bits = LDS byte address
  asm volatile("global_load_async_to_lds_b128 %0, %1, off"
               :: "v"(lds), "v"(g) : "memory");
}
static __device__ __forceinline__ void wait_async0() {
  asm volatile("s_wait_asynccnt 0x0" ::: "memory");
}

// ---------------- elementwise helpers ----------------
__global__ void fill_f32(float* p, float v, int n) {
  int i = blockIdx.x * blockDim.x + threadIdx.x;
  if (i < n) p[i] = v;
}
__global__ void copy_f32(float* d, const float* s, int n) {
  int i = blockIdx.x * blockDim.x + threadIdx.x;
  if (i < n) d[i] = s[i];
}
__global__ void conv_f16(f16* d, const float* s, int n) {
  int i = blockIdx.x * blockDim.x + threadIdx.x;
  if (i < n) d[i] = (f16)s[i];
}
__global__ void add_f32(float* d, const float* s, int n) {
  int i = blockIdx.x * blockDim.x + threadIdx.x;
  if (i < n) d[i] += s[i];
}

// Pre-transpose + convert a weight [K x 128] f32 into f16 chunk images:
// Wt[kc*4096 + n*32 + k] = W[(kc*32+k)*128 + n], zero-padded past K.
__global__ void wt_build(const float* __restrict__ W, f16* __restrict__ Wt,
                         int K, int nchunks) {
  int i = blockIdx.x * blockDim.x + threadIdx.x;
  int total = nchunks * 4096;
  if (i < total) {
    int kc = i >> 12;
    int r  = i & 4095;
    int n  = r >> 5, k = r & 31;
    int gk = kc * 32 + k;
    Wt[i] = (gk < K) ? (f16)W[(size_t)gk * DH + n] : (f16)0.f;
  }
}

// ---------------- graph degree / scatter ----------------
__global__ void deg_count(const int* __restrict__ dstI, float* deg, int E) {
  int i = blockIdx.x * blockDim.x + threadIdx.x;
  if (i < E) atomicAdd(&deg[dstI[i]], 1.0f);
}
__global__ void inv_finalize(float* d, int n) {
  int i = blockIdx.x * blockDim.x + threadIdx.x;
  if (i < n) d[i] = 1.0f / fmaxf(d[i], 1.0f);
}
__global__ void scatter_add(const float* __restrict__ h, const int* __restrict__ srcI,
                            const int* __restrict__ dstI, float* __restrict__ agg,
                            int E, int dim) {
  int gid = blockIdx.x * blockDim.x + threadIdx.x;
  int total = E * dim;
  if (gid < total) {
    int e = gid / dim;
    int f = gid - e * dim;
    atomicAdd(&agg[dstI[e] * dim + f], h[srcI[e] * dim + f]);
  }
}

// ---------------- batchnorm stats ----------------
__global__ void bn_partial(const float* __restrict__ z, float* __restrict__ sums) {
  int f  = threadIdx.x & 127;
  int rl = threadIdx.x >> 7;
  int r0 = blockIdx.x * 256 + rl;
  float s = 0.f, sq = 0.f;
  for (int i = 0; i < 128; ++i) {
    float v = z[(size_t)(r0 + 2 * i) * DH + f];
    s += v; sq += v * v;
  }
  atomicAdd(&sums[f], s);
  atomicAdd(&sums[128 + f], sq);
}
__global__ void bn_finalize(const float* __restrict__ sums, const float* __restrict__ g,
                            const float* __restrict__ bt, float* __restrict__ ab) {
  int f = threadIdx.x;
  float mu  = sums[f] * (1.0f / (float)NNODES);
  float var = sums[128 + f] * (1.0f / (float)NNODES) - mu * mu;
  float a   = g[f] * rsqrtf(var + BN_EPS);
  ab[f] = a;
  ab[128 + f] = bt[f] - mu * a;
}

// ---------------- graph mean pool accumulate ----------------
__global__ void graph_pool_accum(const float* __restrict__ h, float* __restrict__ gp) {
  int b = blockIdx.x, f = threadIdx.x;
  const float* base = h + (size_t)b * NPG * DH + f;
  float s = 0.f;
  for (int i = 0; i < NPG; ++i) s += base[(size_t)i * DH];
  gp[b * DH + f] += s * (1.0f / (float)NPG);
}

// ---------------- generic WMMA GEMM: Z[N x 128] = f(A)[N x K] @ W[K x 128] + bias ----
// Block tile 64 rows x 128 cols; 8 waves, wave = one 16-col tile x 4 m-tiles.
// Weights arrive pre-transposed f16 (Wt); staged into LDS with async b128 copies.
enum { A_PLAIN = 0, A_GINAGG = 1, A_BNRELU = 2, A_TANH = 3, A_AUG = 4 };

template <int MODE>
__global__ void gemm64(const float* __restrict__ A0, const float* __restrict__ A1,
                       const float* __restrict__ A2, const float* __restrict__ A3,
                       const f16* __restrict__ Wt,  const float* __restrict__ bias,
                       float* __restrict__ Z, int K) {
  __shared__ __attribute__((aligned(16))) f16 As[64 * 40];
  __shared__ __attribute__((aligned(16))) f16 Ws[128 * 32];
  const int tid  = threadIdx.x;
  const int row0 = blockIdx.x * 64;
  const int wave = tid >> 5, lane = tid & 31;
  const int mrow = lane & 15, khi = lane >> 4;
  v8f c0 = {}, c1 = {}, c2 = {}, c3 = {};
  for (int ks = 0; ks < K; ks += 32) {
    // async-stage W chunk (4096 halves, contiguous, pre-transposed)
    const f16* gW = Wt + (size_t)(ks >> 5) * 4096;
#pragma unroll
    for (int t = 0; t < 2; ++t) {
      int e = tid + t * 256;
      async_cp16(gW + e * 8, &Ws[e * 8]);
    }
    // stage A tile (64 x 32) with fused pre-op, f32 -> f16 (VALU, overlaps DMA)
#pragma unroll
    for (int t = 0; t < 8; ++t) {
      int e = tid + t * 256;
      int r = e >> 5, k = e & 31;
      int gk = ks + k, grow = row0 + r;
      float val = 0.f;
      if (gk < K) {
        if (MODE == A_PLAIN) {
          val = A0[(size_t)grow * K + gk];
        } else if (MODE == A_GINAGG) {
          val = A0[(size_t)grow * K + gk] + A1[(size_t)grow * K + gk] * A2[grow];
        } else if (MODE == A_BNRELU) {
          float zz = A0[(size_t)grow * K + gk] * A2[gk] + A3[gk];
          val = zz > 0.f ? zz : 0.f;
        } else if (MODE == A_TANH) {
          val = tanhf(A0[(size_t)grow * K + gk]);
        } else { // A_AUG: concat(node_pool, graph_pool[batch])
          val = (gk < DH) ? A0[(size_t)grow * DH + gk]
                          : A1[(size_t)(grow / NPG) * DH + (gk - DH)];
        }
      }
      As[r * 40 + k] = (f16)val;
    }
    if (HAVE_ASYNC) wait_async0();
    __syncthreads();
    // B fragment: lane holds column (wave*16 + mrow), k = 8*khi.. and 16+8*khi..; reused 4x
    int col = wave * 16 + mrow;
    int kb  = khi * 16;
    v16h bf = cat8(*(const v8h*)&Ws[col * 32 + 8 * khi],
                   *(const v8h*)&Ws[col * 32 + 16 + 8 * khi]);
    v16h a0 = cat8(*(const v8h*)&As[(mrow)      * 40 + 8 * khi],
                   *(const v8h*)&As[(mrow)      * 40 + 16 + 8 * khi]);
    v16h a1 = cat8(*(const v8h*)&As[(16 + mrow) * 40 + 8 * khi],
                   *(const v8h*)&As[(16 + mrow) * 40 + 16 + 8 * khi]);
    v16h a2 = cat8(*(const v8h*)&As[(32 + mrow) * 40 + 8 * khi],
                   *(const v8h*)&As[(32 + mrow) * 40 + 16 + 8 * khi]);
    v16h a3 = cat8(*(const v8h*)&As[(48 + mrow) * 40 + 8 * khi],
                   *(const v8h*)&As[(48 + mrow) * 40 + 16 + 8 * khi]);
    c0 = __builtin_amdgcn_wmma_f32_16x16x32_f16(false, a0, false, bf, (short)0, c0, false, false);
    c1 = __builtin_amdgcn_wmma_f32_16x16x32_f16(false, a1, false, bf, (short)0, c1, false, false);
    c2 = __builtin_amdgcn_wmma_f32_16x16x32_f16(false, a2, false, bf, (short)0, c2, false, false);
    c3 = __builtin_amdgcn_wmma_f32_16x16x32_f16(false, a3, false, bf, (short)0, c3, false, false);
    __syncthreads();
  }
  int n   = lane & 15;
  int col = wave * 16 + n;
  float bv = bias[col];
  int mb = khi * 8;
  union { v8f v; float f[8]; } u;
#pragma unroll
  for (int mt = 0; mt < 4; ++mt) {
    u.v = (mt == 0) ? c0 : (mt == 1) ? c1 : (mt == 2) ? c2 : c3;
#pragma unroll
    for (int r = 0; r < 8; ++r)
      Z[(size_t)(row0 + mt * 16 + mb + r) * DH + col] = u.f[r] + bv;
  }
}

// ---------------- pairwise score: S[b] = Hp[b] @ Hp[b]^T, masked -> d_out ----------------
__global__ void score_kernel(const f16* __restrict__ hp, const unsigned char* __restrict__ feas,
                             float* __restrict__ out) {
  __shared__ __attribute__((aligned(16))) f16 As[64 * 40];
  __shared__ __attribute__((aligned(16))) f16 Bs[32 * 40];
  const int bidx = blockIdx.z;
  const int m0 = blockIdx.x * 64, n0 = blockIdx.y * 32;
  const int tid  = threadIdx.x;
  const int wave = tid >> 5, lane = tid & 31;
  const int mt = wave >> 1, nt = wave & 1;
  const int mrow = lane & 15, khi = lane >> 4;
  const f16* base = hp + (size_t)bidx * NPG * DH;
  const int r  = tid >> 2, kv = tid & 3;
  const bool aval = (m0 + r < NPG);
  const bool bpart = (tid < 128);
  const bool bval = bpart && (n0 + r < NPG);
  // pre-zero LDS slots whose rows are out of range (their lanes never overwrite)
  if (!aval) { v8h zz = {}; *(v8h*)&As[r * 40 + kv * 8] = zz; }
  if (bpart && !bval) { v8h zz = {}; *(v8h*)&Bs[r * 40 + kv * 8] = zz; }
  v8f c = {};
  for (int ks = 0; ks < DH; ks += 32) {
    if (aval) async_cp16(&base[(size_t)(m0 + r) * DH + ks + kv * 8], &As[r * 40 + kv * 8]);
    if (bval) async_cp16(&base[(size_t)(n0 + r) * DH + ks + kv * 8], &Bs[r * 40 + kv * 8]);
    if (HAVE_ASYNC) wait_async0();
    __syncthreads();
    v16h af = cat8(*(const v8h*)&As[(mt * 16 + mrow) * 40 + 8 * khi],
                   *(const v8h*)&As[(mt * 16 + mrow) * 40 + 16 + 8 * khi]);
    int kb = khi * 16;
    v16h bf = cat8(*(const v8h*)&Bs[(nt * 16 + mrow) * 40 + 8 * khi],
                   *(const v8h*)&Bs[(nt * 16 + mrow) * 40 + 16 + 8 * khi]);
    c = __builtin_amdgcn_wmma_f32_16x16x32_f16(false, af, false, bf, (short)0, c,
                                               false, false);
    __syncthreads();
  }
  union { v8f v; float f[8]; } u; u.v = c;
  int ng  = n0 + nt * 16 + (lane & 15);
  int mg0 = m0 + mt * 16 + khi * 8;
  if (ng < NPG) {
#pragma unroll
    for (int rr = 0; rr < 8; ++rr) {
      int mg = mg0 + rr;
      if (mg < NPG) {
        size_t idx = (size_t)bidx * NPG * NPG + (size_t)mg * NPG + ng;
        out[idx] = feas[idx] ? u.f[rr] : -__builtin_inff();
      }
    }
  }
}

// ---------------- softmax over flattened 500*500 per batch ----------------
__global__ void softmax_stats(const float* __restrict__ out, float* __restrict__ mx,
                              float* __restrict__ sm) {
  __shared__ float red[512];
  int b = blockIdx.x;
  const float* p = out + (size_t)b * NPG * NPG;
  const int n = NPG * NPG;
  float m = -__builtin_inff();
  for (int i = threadIdx.x; i < n; i += 512) m = fmaxf(m, p[i]);
  red[threadIdx.x] = m; __syncthreads();
  for (int s = 256; s > 0; s >>= 1) {
    if (threadIdx.x < s) red[threadIdx.x] = fmaxf(red[threadIdx.x], red[threadIdx.x + s]);
    __syncthreads();
  }
  m = red[0]; __syncthreads();
  float s = 0.f;
  for (int i = threadIdx.x; i < n; i += 512) s += __expf(p[i] - m);
  red[threadIdx.x] = s; __syncthreads();
  for (int st = 256; st > 0; st >>= 1) {
    if (threadIdx.x < st) red[threadIdx.x] += red[threadIdx.x + st];
    __syncthreads();
  }
  if (threadIdx.x == 0) { mx[b] = m; sm[b] = red[0]; }
}
__global__ void softmax_norm(float* out, const float* __restrict__ mx,
                             const float* __restrict__ sm, int total) {
  int i = blockIdx.x * blockDim.x + threadIdx.x;
  if (i < total) {
    int b = i / (NPG * NPG);
    out[i] = __expf(out[i] - mx[b]) / sm[b];
  }
}

// ---------------- driver ----------------
extern "C" void kernel_launch(void* const* d_in, const int* in_sizes, int n_in,
                              void* d_out, int out_size, void* d_ws, size_t ws_size,
                              hipStream_t stream) {
  const float* x        = (const float*)d_in[0];
  const int*   ei       = (const int*)d_in[1];
  const unsigned char* feas = (const unsigned char*)d_in[3];
  const float* gin0_W1  = (const float*)d_in[4];
  const float* gin0_b1  = (const float*)d_in[5];
  const float* gin0_g1  = (const float*)d_in[6];
  const float* gin0_bt1 = (const float*)d_in[7];
  const float* gin0_W2  = (const float*)d_in[8];
  const float* gin0_b2  = (const float*)d_in[9];
  const float* gin_W1   = (const float*)d_in[10];
  const float* gin_b1   = (const float*)d_in[11];
  const float* gin_g1   = (const float*)d_in[12];
  const float* gin_bt1  = (const float*)d_in[13];
  const float* gin_W2   = (const float*)d_in[14];
  const float* gin_b2   = (const float*)d_in[15];
  const float* p0_W1    = (const float*)d_in[16];
  const float* p0_b1    = (const float*)d_in[17];
  const float* p0_W2    = (const float*)d_in[18];
  const float* p0_b2    = (const float*)d_in[19];
  const float* p_W1     = (const float*)d_in[20];
  const float* p_b1     = (const float*)d_in[21];
  const float* p_W2     = (const float*)d_in[22];
  const float* p_b2     = (const float*)d_in[23];
  float* out = (float*)d_out;

  const int* srcI = ei;
  const int* dstI = ei + NEDGE;

  char* wsb = (char*)d_ws;
  const size_t NF = (size_t)NNODES * DH * sizeof(float);
  float* inv_deg = (float*)wsb; wsb += (size_t)NNODES * sizeof(float);
  float* h       = (float*)wsb; wsb += NF;
  float* agg     = (float*)wsb; wsb += NF;
  float* z       = (float*)wsb; wsb += NF;
  float* npool   = (float*)wsb; wsb += NF;
  float* gpool   = (float*)wsb; wsb += (size_t)NBATCH * DH * sizeof(float);
  float* bn      = (float*)wsb; wsb += 512 * sizeof(float);   // sum|sumsq|a|b
  float* mx      = (float*)wsb; wsb += NBATCH * sizeof(float);
  float* sm      = (float*)wsb; wsb += NBATCH * sizeof(float);
  f16*   hp16    = (f16*)agg;  // reuse after GIN phase

  auto blks = [](long n) { return dim3((unsigned)((n + 255) / 256)); };

  // pre-transposed f16 weight images (chunk-major, zero-padded)
  f16* wt = (f16*)wsb;
  auto CH = [](int K) { return (K + 31) / 32; };
  f16* wt_g0W1 = wt; wt += (size_t)CH(INDIM) * 4096;
  f16* wt_g0W2 = wt; wt += (size_t)CH(DH) * 4096;
  f16* wt_gW1[3]; for (int l = 0; l < 3; ++l) { wt_gW1[l] = wt; wt += (size_t)CH(DH) * 4096; }
  f16* wt_gW2[3]; for (int l = 0; l < 3; ++l) { wt_gW2[l] = wt; wt += (size_t)CH(DH) * 4096; }
  f16* wt_p0W1 = wt; wt += (size_t)CH(2 * DH) * 4096;
  f16* wt_p0W2 = wt; wt += (size_t)CH(DH) * 4096;
  f16* wt_pW1[2]; for (int l = 0; l < 2; ++l) { wt_pW1[l] = wt; wt += (size_t)CH(DH) * 4096; }
  f16* wt_pW2[2]; for (int l = 0; l < 2; ++l) { wt_pW2[l] = wt; wt += (size_t)CH(DH) * 4096; }

  wt_build<<<blks(CH(INDIM) * 4096), 256, 0, stream>>>(gin0_W1, wt_g0W1, INDIM, CH(INDIM));
  wt_build<<<blks(CH(DH) * 4096), 256, 0, stream>>>(gin0_W2, wt_g0W2, DH, CH(DH));
  for (int l = 0; l < 3; ++l) {
    wt_build<<<blks(CH(DH) * 4096), 256, 0, stream>>>(gin_W1 + (size_t)l * DH * DH, wt_gW1[l], DH, CH(DH));
    wt_build<<<blks(CH(DH) * 4096), 256, 0, stream>>>(gin_W2 + (size_t)l * DH * DH, wt_gW2[l], DH, CH(DH));
  }
  wt_build<<<blks(CH(2 * DH) * 4096), 256, 0, stream>>>(p0_W1, wt_p0W1, 2 * DH, CH(2 * DH));
  wt_build<<<blks(CH(DH) * 4096), 256, 0, stream>>>(p0_W2, wt_p0W2, DH, CH(DH));
  for (int l = 0; l < 2; ++l) {
    wt_build<<<blks(CH(DH) * 4096), 256, 0, stream>>>(p_W1 + (size_t)l * DH * DH, wt_pW1[l], DH, CH(DH));
    wt_build<<<blks(CH(DH) * 4096), 256, 0, stream>>>(p_W2 + (size_t)l * DH * DH, wt_pW2[l], DH, CH(DH));
  }

  // degree (self-loop = init 1)
  fill_f32<<<blks(NNODES), 256, 0, stream>>>(inv_deg, 1.0f, NNODES);
  deg_count<<<blks(NEDGE), 256, 0, stream>>>(dstI, inv_deg, NEDGE);
  inv_finalize<<<blks(NNODES), 256, 0, stream>>>(inv_deg, NNODES);

  // ---- GIN layer 0 (in_dim 8 -> 128) ----
  copy_f32<<<blks((long)NNODES * INDIM), 256, 0, stream>>>(agg, x, NNODES * INDIM);
  scatter_add<<<blks((long)NEDGE * INDIM), 256, 0, stream>>>(x, srcI, dstI, agg, NEDGE, INDIM);
  gemm64<A_GINAGG><<<NNODES / 64, 256, 0, stream>>>(x, agg, inv_deg, nullptr, wt_g0W1, gin0_b1, z, INDIM);
  fill_f32<<<1, 256, 0, stream>>>(bn, 0.f, 256);
  bn_partial<<<NNODES / 256, 256, 0, stream>>>(z, bn);
  bn_finalize<<<1, 128, 0, stream>>>(bn, gin0_g1, gin0_bt1, bn + 256);
  gemm64<A_BNRELU><<<NNODES / 64, 256, 0, stream>>>(z, nullptr, bn + 256, bn + 384, wt_g0W2, gin0_b2, h, DH);
  copy_f32<<<blks((long)NNODES * DH), 256, 0, stream>>>(npool, h, NNODES * DH);
  fill_f32<<<blks(NBATCH * DH), 256, 0, stream>>>(gpool, 0.f, NBATCH * DH);
  graph_pool_accum<<<NBATCH, DH, 0, stream>>>(h, gpool);

  // ---- GIN layers 1..3 ----
  for (int l = 0; l < 3; ++l) {
    copy_f32<<<blks((long)NNODES * DH), 256, 0, stream>>>(agg, h, NNODES * DH);
    scatter_add<<<blks((long)NEDGE * DH), 256, 0, stream>>>(h, srcI, dstI, agg, NEDGE, DH);
    gemm64<A_GINAGG><<<NNODES / 64, 256, 0, stream>>>(h, agg, inv_deg, nullptr,
                                                      wt_gW1[l], gin_b1 + l * DH, z, DH);
    fill_f32<<<1, 256, 0, stream>>>(bn, 0.f, 256);
    bn_partial<<<NNODES / 256, 256, 0, stream>>>(z, bn);
    bn_finalize<<<1, 128, 0, stream>>>(bn, gin_g1 + l * DH, gin_bt1 + l * DH, bn + 256);
    gemm64<A_BNRELU><<<NNODES / 64, 256, 0, stream>>>(z, nullptr, bn + 256, bn + 384,
                                                      wt_gW2[l], gin_b2 + l * DH, h, DH);
    add_f32<<<blks((long)NNODES * DH), 256, 0, stream>>>(npool, h, NNODES * DH);
    graph_pool_accum<<<NBATCH, DH, 0, stream>>>(h, gpool);
  }

  // ---- policy MLP ----
  gemm64<A_AUG><<<NNODES / 64, 256, 0, stream>>>(npool, gpool, nullptr, nullptr, wt_p0W1, p0_b1, z, 2 * DH);
  gemm64<A_TANH><<<NNODES / 64, 256, 0, stream>>>(z, nullptr, nullptr, nullptr, wt_p0W2, p0_b2, h, DH);
  for (int l = 0; l < 2; ++l) {
    gemm64<A_PLAIN><<<NNODES / 64, 256, 0, stream>>>(h, nullptr, nullptr, nullptr,
                                                     wt_pW1[l], p_b1 + l * DH, z, DH);
    gemm64<A_TANH><<<NNODES / 64, 256, 0, stream>>>(z, nullptr, nullptr, nullptr,
                                                    wt_pW2[l], p_b2 + l * DH, h, DH);
  }

  // ---- pairwise scores + masked softmax ----
  conv_f16<<<blks((long)NNODES * DH), 256, 0, stream>>>(hp16, h, NNODES * DH);
  score_kernel<<<dim3(8, 16, NBATCH), 256, 0, stream>>>(hp16, feas, out);
  softmax_stats<<<NBATCH, 512, 0, stream>>>(out, mx, sm);
  softmax_norm<<<blks((long)NBATCH * NPG * NPG), 256, 0, stream>>>(out, mx, sm, NBATCH * NPG * NPG);
}
